// MultiHeadAttention4Dtransformer_21320217657944
// MI455X (gfx1250) — matrix operand order repaired and checked by reading, not compiled, not run on hardware
//
#include <hip/hip_runtime.h>

// ---------------------------------------------------------------------------
// Problem constants (from reference): B=4, S=1024, DIM=1024, H=16, DH=64
// ---------------------------------------------------------------------------
#define B_   4
#define S_   1024
#define DIM_ 1024
#define H_   16
#define DH_  64
#define M_   (B_ * S_)   // 4096 rows for the projection GEMMs
#define K_   DIM_        // 1024
#define N_   DIM_        // 1024

typedef __bf16 v16bf __attribute__((ext_vector_type(16)));
typedef float  v8f   __attribute__((ext_vector_type(8)));

#if defined(__HIP_DEVICE_COMPILE__) && __has_builtin(__builtin_amdgcn_tensor_load_to_lds)
#define HAVE_TDM 1
#endif

// ---------------------------------------------------------------------------
// WMMA helper: D = A(16x32 bf16) * B(32x16 bf16) + C(16x16 f32)
// ---------------------------------------------------------------------------
static __device__ inline v8f wmma_bf16(v16bf a, v16bf b, v8f c) {
  return __builtin_amdgcn_wmma_f32_16x16x32_bf16(
      /*neg_a=*/false, a, /*neg_b=*/false, b,
      /*c_mod=*/(short)0, c, /*reuse_a=*/false, /*reuse_b=*/false);
}

// A-matrix fragment (16x32 bf16):
//   lanes 0-15 : row M=lane,    halves = K 0..15
//   lanes 16-31: row M=lane-16, halves = K 16..31
static __device__ inline v16bf a_frag(const __bf16* base, int lane, int stride) {
  const int row  = lane & 15;
  const int kofs = (lane >> 4) << 4;  // 0 or 16
  const __bf16* p = base + (size_t)row * stride + kofs;
  v16bf a;
#pragma unroll
  for (int i = 0; i < 16; ++i) a[i] = p[i];
  return a;
}

// B-matrix fragment (32x16 bf16):
//   lanes 0-15 : col N=lane,    halves = K 0..15
//   lanes 16-31: col N=lane-16, halves = K 16..31
// element (kk,n) at base[kk*kstride + n]  (K-major storage)
static __device__ inline v16bf b_frag_kmajor(const __bf16* base, int lane, int kstride) {
  const int n  = lane & 15;
  const int k0 = (lane >> 4) << 4;  // 0 or 16
  v16bf b;
#pragma unroll
  for (int i = 0; i < 16; ++i) b[i] = base[(size_t)(k0 + i) * kstride + n];
  return b;
}

// element (kk,n) at base[n*nstride + kk]  (N-major storage, K contiguous)
static __device__ inline v16bf b_frag_nmajor(const __bf16* base, int lane, int nstride) {
  const int n  = lane & 15;
  const int k0 = (lane >> 4) << 4;
  const __bf16* p = base + (size_t)n * nstride + k0;
  v16bf b;
#pragma unroll
  for (int i = 0; i < 16; ++i) b[i] = p[i];
  return b;
}

// Pre-packed fragment layout: each lane's 16 halves contiguous (32B -> 2x b128)
static __device__ inline v16bf b_frag_packed(const __bf16* base, int lane) {
  const __bf16* p = base + (lane << 4);
  v16bf b;
#pragma unroll
  for (int i = 0; i < 16; ++i) b[i] = p[i];
  return b;
}

// ---------------------------------------------------------------------------
// wave32 reductions / scan (gfx1250 is wave32-only)
// ---------------------------------------------------------------------------
static __device__ inline float wave_max(float v) {
#pragma unroll
  for (int o = 16; o > 0; o >>= 1) v = fmaxf(v, __shfl_xor(v, o, 32));
  return v;
}
static __device__ inline float wave_sum(float v) {
#pragma unroll
  for (int o = 16; o > 0; o >>= 1) v += __shfl_xor(v, o, 32);
  return v;
}
static __device__ inline float wave_scan_incl(float v, int lane) {
#pragma unroll
  for (int o = 1; o < 32; o <<= 1) {
    float t = __shfl_up(v, o, 32);
    if (lane >= o) v += t;
  }
  return v;
}

// ---------------------------------------------------------------------------
// Tensor Data Mover: DMA a (rows x 64) bf16 tile, row stride 64, into LDS.
// D# per ISA cdna5 §8: group0 {count, lds_addr, global_addr, type=2},
// group1 {data_size=1(2B), tensor_dim0=64, tensor_dim1=1024, tile_dim0=64,
//         tile_dim1=rows, tensor_dim0_stride=64}. Groups 2/3 unused (2D).
// lds_addr assumes dynamic-LDS base at offset 0 of the workgroup allocation.
// This toolchain exposes the 6-arg builtin form:
//   (u32x4 g0, i32x8 g1, i32x4 g2, i32x4 g3, i32x8 extra, i32 cpol)
// ---------------------------------------------------------------------------
#ifdef HAVE_TDM
typedef unsigned int u32x4 __attribute__((ext_vector_type(4)));
typedef int          i32x4 __attribute__((ext_vector_type(4)));
typedef int          i32x8 __attribute__((ext_vector_type(8)));

static __device__ inline void tdm_load_tile64(const __bf16* gsrc, unsigned lds_off,
                                              int rows) {
  const unsigned long long ga = (unsigned long long)(const void*)gsrc;
  u32x4 g0;
  g0[0] = 1u;                                      // count=1 valid user D#
  g0[1] = lds_off;                                 // lds_addr (bytes)
  g0[2] = (unsigned)(ga & 0xffffffffu);            // global_addr[31:0]
  g0[3] = (unsigned)((ga >> 32) & 0x01ffffffu)     // global_addr[56:32]
          | (2u << 30);                            // type = 2 ("image")
  i32x8 g1;
  g1[0] = (1 << 16);        // workgroup_mask=0 | data_size=1 (2 bytes)
  g1[1] = (64 << 16);       // tensor_dim0[15:0]=64 in bits 63:48
  g1[2] = (1024 << 16);     // tensor_dim0 hi=0 | tensor_dim1[15:0]=1024
  g1[3] = (64 << 16);       // tensor_dim1 hi=0 | tile_dim0=64
  g1[4] = rows & 0xffff;    // tile_dim1=rows | tile_dim2=0
  g1[5] = 64;               // tensor_dim0_stride[31:0]=64
  g1[6] = 0;                // stride hi | tensor_dim1_stride lo
  g1[7] = 0;
  const i32x4 z4 = {0, 0, 0, 0};
  const i32x8 z8 = {0, 0, 0, 0, 0, 0, 0, 0};
  __builtin_amdgcn_tensor_load_to_lds(g0, g1, z4, z4, z8, 0);
}
#endif

// ---------------------------------------------------------------------------
// Projection GEMM:  C[m,n] = sum_k X[m,k] * W[n,k] + bias[n]
//   X: (4096,1024) TIN (fp32 or bf16),  W: (1024,1024) fp32 row-major
//   HEAD_OUT=1: write bf16 to head-major (B,H,S,DH); else fp32 to (M,N)
// Block: 128 threads (4 waves), 64x64 output tile, K-step 32 through LDS.
// W is staged directly in B-fragment order so each lane reads its whole
// fragment as 32 contiguous bytes (2 x ds_load_b128, conflict-free).
// ---------------------------------------------------------------------------
template <typename TIN, bool HEAD_OUT>
__global__ __launch_bounds__(128) void proj_gemm(const TIN* __restrict__ X,
                                                 const float* __restrict__ W,
                                                 const float* __restrict__ bias,
                                                 void* __restrict__ outp) {
  __shared__ __bf16 Xs[64 * 32];      // [m][k], stride 32
  __shared__ __bf16 Ws[4 * 32 * 16];  // [nc][lane][half] packed B fragments

  const int n0 = blockIdx.x * 64;
  const int m0 = blockIdx.y * 64;
  const int t    = threadIdx.x;
  const int wave = t >> 5;
  const int lane = t & 31;

  v8f acc[4];
#pragma unroll
  for (int c = 0; c < 4; ++c)
#pragma unroll
    for (int i = 0; i < 8; ++i) acc[c][i] = 0.0f;

  for (int k0 = 0; k0 < K_; k0 += 32) {
    // Stage: coalesced global reads, fp32 -> bf16 convert into LDS.
    for (int idx = t; idx < 2048; idx += 128) {
      const int r = idx >> 5;   // 0..63  (m row / n row)
      const int c = idx & 31;   // 0..31  (k)
      Xs[idx] = (__bf16)(float)X[(size_t)(m0 + r) * K_ + k0 + c];
      // W element (n=r, kk=c) scattered into packed B-fragment order:
      //   slot = nc*32 + (n&15 | (kk&16)),  half = kk&15
      const int slot = ((r >> 4) << 5) + ((r & 15) | (c & 16));
      Ws[(slot << 4) + (c & 15)] = (__bf16)W[(size_t)(n0 + r) * K_ + k0 + c];
    }
    // Prefetch next K-step tiles into cache while WMMAs run.
    if (k0 + 32 < K_) {
      __builtin_prefetch(&X[(size_t)(m0 + (t & 63)) * K_ + k0 + 32], 0, 0);
      __builtin_prefetch(&W[(size_t)(n0 + (t & 63)) * K_ + k0 + 32], 0, 0);
    }
    __syncthreads();

    const v16bf a = a_frag(Xs + wave * 16 * 32, lane, 32);
#pragma unroll
    for (int nc = 0; nc < 4; ++nc) {
      const v16bf bf = b_frag_packed(Ws + nc * 512, lane);
      acc[nc] = wmma_bf16(a, bf, acc[nc]);
    }
    __syncthreads();
  }

  // C layout: lanes 0-15 -> M=r, N=lane; lanes 16-31 -> M=8+r, N=lane-16
  const int ml = (lane >> 4) << 3;
#pragma unroll
  for (int nc = 0; nc < 4; ++nc) {
#pragma unroll
    for (int r = 0; r < 8; ++r) {
      const int m = m0 + wave * 16 + ml + r;
      const int n = n0 + nc * 16 + (lane & 15);
      const float val = acc[nc][r] + bias[n];
      if (HEAD_OUT) {
        const int b = m >> 10, s = m & 1023;     // m = b*S + s
        const int h = n >> 6,  d = n & 63;       // n = h*DH + d
        ((__bf16*)outp)[((size_t)(b * H_ + h) * S_ + s) * DH_ + d] = (__bf16)val;
      } else {
        ((float*)outp)[(size_t)m * N_ + n] = val;
      }
    }
  }
}

// ---------------------------------------------------------------------------
// Attention: one wave per (b, h, 16-row query tile).
//   scores = QK^T/8 (WMMA, causal) -> softmax -> cumsum-based distance decay
//   -> second softmax -> scores to d_out (fp32), P@V (WMMA) -> concat bf16.
// LDS: [0,64K) fp32 scores, [64K,96K) bf16 probs, [96K,104K) V double buffer
// (TDM destination). V tiles are DMA'd by the Tensor Data Mover one tile
// ahead of the WMMA consumer, synchronized with s_wait_tensorcnt.
// ---------------------------------------------------------------------------
#define LDS_SC_OFF 0
#define LDS_PB_OFF (16 * 1024 * 4)                   // 64 KB
#define LDS_VS_OFF (LDS_PB_OFF + 16 * 1024 * 2)      // 96 KB
#define LDS_TOTAL  (LDS_VS_OFF + 2 * 32 * 64 * 2)    // 104 KB

__global__ __launch_bounds__(32) void attn_kernel(const __bf16* __restrict__ Qh,
                                                  const __bf16* __restrict__ Kh,
                                                  const __bf16* __restrict__ Vh,
                                                  const float* __restrict__ gammas,
                                                  float* __restrict__ scores_out,
                                                  __bf16* __restrict__ concat) {
  extern __shared__ char smem[];
  float*  sc = (float*)(smem + LDS_SC_OFF);   // [16][1024] fp32
  __bf16* pb = (__bf16*)(smem + LDS_PB_OFF);  // [16][1024] bf16

  const int bh   = blockIdx.x >> 6;   // 0..63  (b*H + h)
  const int qt   = blockIdx.x & 63;   // query tile index
  const int q0   = qt * 16;
  const int lane = threadIdx.x;
  const int b    = bh >> 4;
  const int h    = bh & 15;

  const __bf16* Q = Qh + (size_t)bh * S_ * DH_;
  const __bf16* K = Kh + (size_t)bh * S_ * DH_;
  const __bf16* V = Vh + (size_t)bh * S_ * DH_;
  const float g = -fabsf(gammas[h]);

  // ---- scores tile row: S = Q(16x64) @ K^T, causal masked ----
  const v16bf aq0 = a_frag(Q + (size_t)q0 * DH_,      lane, DH_);  // dh 0..31
  const v16bf aq1 = a_frag(Q + (size_t)q0 * DH_ + 32, lane, DH_);  // dh 32..63
  const int ml = (lane >> 4) << 3;
  for (int jt = 0; jt <= qt; ++jt) {
    v8f acc;
#pragma unroll
    for (int i = 0; i < 8; ++i) acc[i] = 0.0f;
    const v16bf bk0 = b_frag_nmajor(K + (size_t)jt * 16 * DH_,      lane, DH_);
    const v16bf bk1 = b_frag_nmajor(K + (size_t)jt * 16 * DH_ + 32, lane, DH_);
    acc = wmma_bf16(aq0, bk0, acc);
    acc = wmma_bf16(aq1, bk1, acc);
    const int jj = jt * 16 + (lane & 15);
#pragma unroll
    for (int r = 0; r < 8; ++r) {
      const int i = q0 + ml + r;
      sc[(ml + r) * 1024 + jj] = (jj <= i) ? acc[r] * 0.125f : -1.0e32f;
    }
  }
  __syncthreads();

  // ---- per-row: softmax -> cumsum decay -> second softmax ----
  for (int il = 0; il < 16; ++il) {
    const int i  = q0 + il;
    const int nj = i + 1;  // causal: valid cols j in [0, i]
    float* row = sc + il * 1024;

    float m1 = -3.4e38f;
    for (int j = lane; j < nj; j += 32) m1 = fmaxf(m1, row[j]);
    m1 = wave_max(m1);

    float se = 0.0f;
    for (int j = lane; j < nj; j += 32) se += __expf(row[j] - m1);
    se = wave_sum(se);
    const float inv_se = 1.0f / se;  // disttot == se (p sums to 1)

    float run = 0.0f, m2 = -3.4e38f;
    for (int jb = 0; jb < nj; jb += 32) {
      const int j = jb + lane;
      const float e  = (j < nj) ? __expf(row[j] - m1) : 0.0f;
      const float cs = wave_scan_incl(e, lane);
      const float cume = run + cs;
      run += __shfl(cs, 31, 32);  // block total, uniform
      if (j < nj) {
        const float rem  = 1.0f - cume * inv_se;          // disttot - distcum
        const float dist = sqrtf(fmaxf(rem * (float)(i - j), 0.0f));
        float eff = __expf(g * dist);
        eff = fminf(fmaxf(eff, 1e-5f), 1e5f);
        const float s2 = row[j] * eff;
        row[j] = s2;
        m2 = fmaxf(m2, s2);
      }
    }
    m2 = wave_max(m2);

    float s2s = 0.0f;
    for (int j = lane; j < nj; j += 32) {
      const float e2 = __expf(row[j] - m2);
      row[j] = e2;
      s2s += e2;
    }
    s2s = wave_sum(s2s);
    const float inv2 = 1.0f / s2s;

    float*  orow = scores_out + ((size_t)bh * S_ + i) * S_;
    __bf16* prow = pb + il * 1024;
    for (int j = lane; j < S_; j += 32) {
      const float p = (j < nj) ? row[j] * inv2 : 0.0f;
      orow[j] = p;
      prow[j] = (__bf16)p;
    }
  }
  __syncthreads();

  // ---- out = P(16x1024 bf16) @ V(1024x64), causal-bounded K range ----
  v8f oacc[4];
#pragma unroll
  for (int c = 0; c < 4; ++c)
#pragma unroll
    for (int i = 0; i < 8; ++i) oacc[c][i] = 0.0f;

  const int kbmax = (q0 + 16 + 31) >> 5;  // cover j <= q0+15

#ifdef HAVE_TDM
  __bf16* vstage = (__bf16*)(smem + LDS_VS_OFF);  // [2][32*64] bf16
  // Prologue: DMA first V tile into buffer 0.
  tdm_load_tile64(V, LDS_VS_OFF, 32);
#endif

  for (int kb = 0; kb < kbmax; ++kb) {
#ifdef HAVE_TDM
    if (kb + 1 < kbmax) {
      // Issue DMA for the next tile, then wait until only it is in flight
      // (i.e. the tile for this iteration has landed in LDS).
      tdm_load_tile64(V + (size_t)(kb + 1) * 32 * DH_,
                      LDS_VS_OFF + ((kb + 1) & 1) * (32 * 64 * 2), 32);
      __builtin_amdgcn_s_wait_tensorcnt(1);
    } else {
      __builtin_amdgcn_s_wait_tensorcnt(0);
    }
    const __bf16* vt = vstage + (kb & 1) * (32 * 64);
#else
    const __bf16* vt = V + (size_t)kb * 32 * DH_;
#endif
    const v16bf ap = a_frag(pb + kb * 32, lane, 1024);
#pragma unroll
    for (int dc = 0; dc < 4; ++dc) {
      // element (kk,n) = vt[kk*64 + dc*16 + n]
      const v16bf bv = b_frag_kmajor(vt + dc * 16, lane, DH_);
      oacc[dc] = wmma_bf16(ap, bv, oacc[dc]);
    }
  }

#pragma unroll
  for (int dc = 0; dc < 4; ++dc) {
#pragma unroll
    for (int r = 0; r < 8; ++r) {
      const int s = q0 + ml + r;
      const int d = dc * 16 + (lane & 15);
      concat[((size_t)b * S_ + s) * DIM_ + h * DH_ + d] = (__bf16)oacc[dc][r];
    }
  }
}

// ---------------------------------------------------------------------------
// Launch: 3 projections -> attention -> output projection (all on `stream`)
// Workspace: qh, kh, vh (head-major bf16) + concat bf16  = 32 MB total
// d_out: [output (4,1024,1024) fp32][scores (4,16,1024,1024) fp32]
// ---------------------------------------------------------------------------
extern "C" void kernel_launch(void* const* d_in, const int* in_sizes, int n_in,
                              void* d_out, int out_size, void* d_ws, size_t ws_size,
                              hipStream_t stream) {
  const float* q      = (const float*)d_in[0];
  const float* k      = (const float*)d_in[1];
  const float* v      = (const float*)d_in[2];
  // d_in[3] = mask (causal, reconstructed analytically in-kernel)
  const float* Wq     = (const float*)d_in[4];
  const float* bq     = (const float*)d_in[5];
  const float* Wk     = (const float*)d_in[6];
  const float* bk     = (const float*)d_in[7];
  const float* Wv     = (const float*)d_in[8];
  const float* bv     = (const float*)d_in[9];
  const float* Wo     = (const float*)d_in[10];
  const float* bo     = (const float*)d_in[11];
  const float* gammas = (const float*)d_in[12];

  const size_t HEAD_ELEMS = (size_t)B_ * H_ * S_ * DH_;  // 4M
  __bf16* qh     = (__bf16*)d_ws;
  __bf16* kh     = qh + HEAD_ELEMS;
  __bf16* vh     = kh + HEAD_ELEMS;
  __bf16* concat = vh + HEAD_ELEMS;

  float* out    = (float*)d_out;
  float* scores = out + (size_t)B_ * S_ * DIM_;

  const dim3 pgrid(N_ / 64, M_ / 64);  // (16, 64)
  proj_gemm<float, true><<<pgrid, 128, 0, stream>>>(q, Wq, bq, qh);
  proj_gemm<float, true><<<pgrid, 128, 0, stream>>>(k, Wk, bk, kh);
  proj_gemm<float, true><<<pgrid, 128, 0, stream>>>(v, Wv, bv, vh);

  attn_kernel<<<dim3(B_ * H_ * (S_ / 16)), 32, LDS_TOTAL, stream>>>(
      qh, kh, vh, gammas, scores, concat);

  proj_gemm<__bf16, false><<<pgrid, 128, 0, stream>>>(concat, Wo, bo, out);
}
